// MultiHeadedAttention_9706626089976
// MI455X (gfx1250) — compile-verified, matrix operand order checked by reading, not run on hardware
//
#include <hip/hip_runtime.h>
#include <hip/hip_bf16.h>
#include <cmath>

typedef __bf16 bf16;
typedef __attribute__((ext_vector_type(16))) __bf16 v16bf;
typedef __attribute__((ext_vector_type(8)))  __bf16 v8bf;
typedef __attribute__((ext_vector_type(8)))  float  v8f;

#define BT    16
#define CH    768
#define HWN   4096
#define WW    64
#define BB    2
#define TT    8
#define DK    256
#define NDPB  8388608      /* n*d per (scale,b) = T*DK*64*64 */
#define KCONV 6912         /* 768*9 */

#define BM 128
#define BN 128
#define BK 32
#define LDA 40             /* [row][k] tile stride (bf16): 80B rows, 16B aligned */
#define NTHREADS 256

// ---- fragment from [row][k] LDS tile: two contiguous 16B LDS loads ----
// CDNA5 16-bit layout: lane<16 holds K {0..7,16..23}, lane>=16 {8..15,24..31}
__device__ __forceinline__ v16bf frag_rowk(const bf16* base, int lane) {
    const bf16* p = base + (lane & 15) * LDA + ((lane < 16) ? 0 : 8);
    union { v16bf v; v8bf h[2]; } f;
    f.h[0] = *(const v8bf*)p;          // k {0..7}   (or {8..15})
    f.h[1] = *(const v8bf*)(p + 16);   // k {16..23} (or {24..31})
    return f.v;
}

__device__ __forceinline__ v8f wmma_bf16(v16bf a, v16bf b, v8f c) {
    return __builtin_amdgcn_wmma_f32_16x16x32_bf16(false, a, false, b, (short)0, c, false, false);
}

// async global -> LDS 16B copy per lane (tracked by ASYNCcnt)
__device__ __forceinline__ void async_cp16(const bf16* gptr, bf16* lptr) {
    unsigned lds = (unsigned)(unsigned long long)(uintptr_t)lptr; // LDS offset = low 32 bits
    asm volatile("global_load_async_to_lds_b128 %0, %1, off"
                 :: "v"(lds), "v"(gptr)
                 : "memory");
}
__device__ __forceinline__ void async_wait0() {
    asm volatile("s_wait_asynccnt 0x0" ::: "memory");
}

// 8-wmma compute step shared by all GEMMs (wave = 32x64 slab of 128x128 block)
#define COMPUTE_STEP(AS, BS)                                                  \
    {                                                                         \
        v16bf a0 = frag_rowk(&AS[(wm * 32) * LDA], lane);                     \
        v16bf a1 = frag_rowk(&AS[(wm * 32 + 16) * LDA], lane);                \
        _Pragma("unroll")                                                     \
        for (int jn = 0; jn < 4; ++jn) {                                      \
            v16bf bb = frag_rowk(&BS[(wn * 64 + jn * 16) * LDA], lane);       \
            acc[0][jn] = wmma_bf16(a0, bb, acc[0][jn]);                       \
            acc[1][jn] = wmma_bf16(a1, bb, acc[1][jn]);                       \
        }                                                                     \
    }

// =====================================================================
// Kernel 1: QKV 1x1-conv GEMM + windowed-pack epilogue.
// Q/K packed [n][d]; V packed transposed [d][n] (feeds y-GEMM's B side).
// grid: (CH/BM, HWN/BN, BT*3)
// =====================================================================
__global__ __launch_bounds__(NTHREADS)
void qkv_gemm_pack(const float* __restrict__ x,
                   const float* __restrict__ wq, const float* __restrict__ bq,
                   const float* __restrict__ wk, const float* __restrict__ bk,
                   const float* __restrict__ wv, const float* __restrict__ bv,
                   bf16* __restrict__ qpack, bf16* __restrict__ kpack,
                   bf16* __restrict__ vpackT)
{
    __shared__ __align__(16) bf16 As[2][BM * LDA];
    __shared__ __align__(16) bf16 Bs[2][BN * LDA];
    int tid = threadIdx.x, lane = tid & 31, wid = tid >> 5;
    int wm = wid >> 1, wn = wid & 1;
    int m0 = blockIdx.x * BM;
    int n0 = blockIdx.y * BN;
    int z  = blockIdx.z;
    int bt = z / 3, which = z % 3;
    const float* wmat = (which == 0) ? wq : (which == 1) ? wk : wv;
    const float* bias = (which == 0) ? bq : (which == 1) ? bk : bv;
    bf16* outp        = (which == 0) ? qpack : (which == 1) ? kpack : vpackT;
    const float* xb = x + (size_t)bt * CH * HWN;

    // A: weights, rows contiguous along k -> straight octet copy+convert
    auto stageA = [&](int k0, int b) {
#pragma unroll
        for (int j = 0; j < 2; ++j) {
            int c = j * NTHREADS + tid;            // 512 chunks of 8 along k
            int mr = c >> 2, kq = c & 3;
            const float* s = &wmat[(size_t)(m0 + mr) * CH + k0 + kq * 8];
            float4 f0 = *(const float4*)s, f1 = *(const float4*)(s + 4);
            v8bf h = {(bf16)f0.x, (bf16)f0.y, (bf16)f0.z, (bf16)f0.w,
                      (bf16)f1.x, (bf16)f1.y, (bf16)f1.z, (bf16)f1.w};
            *(v8bf*)&As[b][mr * LDA + kq * 8] = h;
        }
    };
    // B: x is [k][n]; transpose at stage: 8 k-strided loads (coalesced in n),
    // pack, one 16B LDS store into [col][k] tile.
    auto stageB = [&](int k0, int b) {
#pragma unroll
        for (int j = 0; j < 2; ++j) {
            int c = j * NTHREADS + tid;            // 512 chunks: col x k-octet
            int kq = c >> 7, col = c & 127;
            const float* s = &xb[(size_t)(k0 + kq * 8) * HWN + n0 + col];
            float t[8];
#pragma unroll
            for (int e = 0; e < 8; ++e) t[e] = s[(size_t)e * HWN];
            v8bf h = {(bf16)t[0], (bf16)t[1], (bf16)t[2], (bf16)t[3],
                      (bf16)t[4], (bf16)t[5], (bf16)t[6], (bf16)t[7]};
            *(v8bf*)&Bs[b][col * LDA + kq * 8] = h;
        }
    };

    v8f acc[2][4] = {};
    const int steps = CH / BK;
    stageA(0, 0); stageB(0, 0);
    for (int i = 0; i < steps; ++i) {
        __syncthreads();
        if (i + 1 < steps) { stageA((i + 1) * BK, (i + 1) & 1); stageB((i + 1) * BK, (i + 1) & 1); }
        int cb = i & 1;
        COMPUTE_STEP(As[cb], Bs[cb]);
    }

    int cn = lane & 15, mrow0 = (lane < 16) ? 0 : 8;
    int b_idx = bt >> 3, ti = bt & 7;
#pragma unroll
    for (int im = 0; im < 2; ++im)
#pragma unroll
    for (int jn = 0; jn < 4; ++jn) {
        int ncol = n0 + wn * 64 + jn * 16 + cn;
        int yy = ncol >> 6, xx = ncol & 63;
#pragma unroll
        for (int r = 0; r < 8; ++r) {
            int o = m0 + wm * 32 + im * 16 + mrow0 + r;
            float v = acc[im][jn][r] + bias[o];
            int scale = o >> 8, ci = o & 255;
            int lp = 4 - scale;                 // log2(patch)
            int so = scale + 2;                 // log2(op)
            int pm = (1 << lp) - 1;
            int nwin = (ti << (2 * so)) + ((yy >> lp) << so) + (xx >> lp);
            int d    = (ci << (2 * lp)) + ((yy & pm) << lp) + (xx & pm);
            size_t base = ((size_t)scale * BB + b_idx) * (size_t)NDPB;
            size_t dst;
            if (which == 2) {                   // V stored transposed [d][n]
                int Nw = TT << (2 * so);
                dst = base + (size_t)d * Nw + nwin;
            } else {                            // Q/K stored [n][d]
                size_t Dd = (size_t)DK << (2 * lp);
                dst = base + (size_t)nwin * Dd + d;
            }
            outp[dst] = (bf16)v;
        }
    }
}

// =====================================================================
// Kernel 2: scores = sfac * qq . kk^T  (pure bf16 copies -> async DMA to LDS)
// grid: (Nwin/BM, Nwin/BN, BB*nchunks)
// =====================================================================
__global__ __launch_bounds__(NTHREADS)
void scores_gemm(const bf16* __restrict__ qpack, const bf16* __restrict__ kpack,
                 float* __restrict__ scores,
                 int scale, int Nwin, int Dd, int KC, int nchunks, float sfac)
{
    __shared__ __align__(16) bf16 As[2][BM * LDA];
    __shared__ __align__(16) bf16 Bt[2][BN * LDA];
    int tid = threadIdx.x, lane = tid & 31, wid = tid >> 5;
    int wm = wid >> 1, wn = wid & 1;
    int m0 = blockIdx.x * BM;
    int n0 = blockIdx.y * BN;
    int z  = blockIdx.z;
    int b_idx = z / nchunks, chunk = z % nchunks;
    const bf16* qb = qpack + ((size_t)scale * BB + b_idx) * (size_t)NDPB;
    const bf16* kb = kpack + ((size_t)scale * BB + b_idx) * (size_t)NDPB;

    auto stage = [&](const bf16* src, int r0, bf16* dstTile, int k0) {
#pragma unroll
        for (int j = 0; j < 2; ++j) {
            int c = j * NTHREADS + tid;
            int mr = c >> 2, kq = c & 3;
            async_cp16(&src[(size_t)(r0 + mr) * Dd + k0 + kq * 8],
                       &dstTile[mr * LDA + kq * 8]);
        }
    };

    v8f acc[2][4] = {};
    int kbeg = chunk * KC;
    int kend = kbeg + KC; if (kend > Dd) kend = Dd;
    const int steps = (kend - kbeg) / BK;
    stage(qb, m0, As[0], kbeg); stage(kb, n0, Bt[0], kbeg);
    for (int i = 0; i < steps; ++i) {
        async_wait0();
        __syncthreads();
        if (i + 1 < steps) {
            int nk = kbeg + (i + 1) * BK, nb = (i + 1) & 1;
            stage(qb, m0, As[nb], nk); stage(kb, n0, Bt[nb], nk);
        }
        int cb = i & 1;
        COMPUTE_STEP(As[cb], Bt[cb]);
    }

    float* sb = scores + (size_t)b_idx * Nwin * Nwin;
    int cn = lane & 15, mrow0 = (lane < 16) ? 0 : 8;
#pragma unroll
    for (int im = 0; im < 2; ++im)
#pragma unroll
    for (int jn = 0; jn < 4; ++jn) {
        int col = n0 + wn * 64 + jn * 16 + cn;
#pragma unroll
        for (int r = 0; r < 8; ++r) {
            int row = m0 + wm * 32 + im * 16 + mrow0 + r;
            float v = acc[im][jn][r] * sfac;
            if (nchunks == 1) sb[(size_t)row * Nwin + col] = v;
            else atomicAdd(&sb[(size_t)row * Nwin + col], v);
        }
    }
}

__global__ void zero_f32(float* __restrict__ p, int n) {
    int i = blockIdx.x * blockDim.x + threadIdx.x;
    if (i < n) p[i] = 0.0f;
}

// =====================================================================
// Kernel 3: row softmax, fp32 -> bf16.  grid: (BB*Nwin)
// =====================================================================
__global__ __launch_bounds__(256)
void softmax_rows(const float* __restrict__ scores, bf16* __restrict__ pmat, int Nwin)
{
    __shared__ float red[256];
    int row = blockIdx.x;
    const float* s = scores + (size_t)row * Nwin;
    bf16* o = pmat + (size_t)row * Nwin;
    int tid = threadIdx.x;
    float mx = -1e30f;
    for (int i = tid; i < Nwin; i += 256) mx = fmaxf(mx, s[i]);
    red[tid] = mx; __syncthreads();
    for (int st = 128; st > 0; st >>= 1) {
        if (tid < st) red[tid] = fmaxf(red[tid], red[tid + st]);
        __syncthreads();
    }
    mx = red[0]; __syncthreads();
    float sum = 0.f;
    for (int i = tid; i < Nwin; i += 256) sum += __expf(s[i] - mx);
    red[tid] = sum; __syncthreads();
    for (int st = 128; st > 0; st >>= 1) {
        if (tid < st) red[tid] += red[tid + st];
        __syncthreads();
    }
    float inv = 1.0f / red[0];
    for (int i = tid; i < Nwin; i += 256) o[i] = (bf16)(__expf(s[i] - mx) * inv);
}

// =====================================================================
// Kernel 4: y = p . vv with vv pre-transposed [d][m] -> both stages are
// async DMA copies. Epilogue un-windows into bf16 NCHW attn (shift math).
// grid: (Nwin/BM, Dd/BN, BB)
// =====================================================================
template <int SCALE>
__global__ __launch_bounds__(NTHREADS)
void y_gemm_unpack(const bf16* __restrict__ pmat, const bf16* __restrict__ vpackT,
                   bf16* __restrict__ attn)
{
    constexpr int LP   = 4 - SCALE;            // log2(patch)
    constexpr int SO   = SCALE + 2;            // log2(op)
    constexpr int P    = 1 << LP;
    constexpr int OP   = 1 << SO;
    constexpr int Nwin = TT * OP * OP;
    constexpr int Dd   = DK * P * P;

    __shared__ __align__(16) bf16 As[2][BM * LDA];
    __shared__ __align__(16) bf16 Bt[2][BN * LDA];
    int tid = threadIdx.x, lane = tid & 31, wid = tid >> 5;
    int wm = wid >> 1, wn = wid & 1;
    int m0 = blockIdx.x * BM;      // window rows
    int n0 = blockIdx.y * BN;      // d columns
    int b_idx = blockIdx.z;
    const bf16* pb = pmat + (size_t)b_idx * Nwin * Nwin;
    const bf16* vb = vpackT + ((size_t)SCALE * BB + b_idx) * (size_t)NDPB;

    auto stage = [&](const bf16* src, int srcld, int r0, bf16* dstTile, int k0) {
#pragma unroll
        for (int j = 0; j < 2; ++j) {
            int c = j * NTHREADS + tid;
            int mr = c >> 2, kq = c & 3;
            async_cp16(&src[(size_t)(r0 + mr) * srcld + k0 + kq * 8],
                       &dstTile[mr * LDA + kq * 8]);
        }
    };

    v8f acc[2][4] = {};
    const int steps = Nwin / BK;
    stage(pb, Nwin, m0, As[0], 0); stage(vb, Nwin, n0, Bt[0], 0);
    for (int i = 0; i < steps; ++i) {
        async_wait0();
        __syncthreads();
        if (i + 1 < steps) {
            int nk = (i + 1) * BK, nb = (i + 1) & 1;
            stage(pb, Nwin, m0, As[nb], nk); stage(vb, Nwin, n0, Bt[nb], nk);
        }
        int cb = i & 1;
        COMPUTE_STEP(As[cb], Bt[cb]);
    }

    int cn = lane & 15, mrow0 = (lane < 16) ? 0 : 8;
#pragma unroll
    for (int im = 0; im < 2; ++im)
#pragma unroll
    for (int jn = 0; jn < 4; ++jn) {
        int d  = n0 + wn * 64 + jn * 16 + cn;
        int ci = d >> (2 * LP);
        int py = (d >> LP) & (P - 1), px = d & (P - 1);
#pragma unroll
        for (int r = 0; r < 8; ++r) {
            int nwin = m0 + wm * 32 + im * 16 + mrow0 + r;
            int ti  = nwin >> (2 * SO);
            int ohi = (nwin >> SO) & (OP - 1), owi = nwin & (OP - 1);
            int bt = b_idx * TT + ti;
            int o  = SCALE * DK + ci;
            int yy = (ohi << LP) + py, xx = (owi << LP) + px;
            attn[((size_t)bt * CH + o) * HWN + yy * WW + xx] = (bf16)acc[im][jn][r];
        }
    }
}

// =====================================================================
// Kernel 5: 3x3 conv implicit GEMM (K=6912) + bias + LeakyReLU(0.2)
// grid: (CH/BM, HWN/BN, BT)
// =====================================================================
__global__ __launch_bounds__(NTHREADS)
void conv3x3_gemm(const bf16* __restrict__ attn, const float* __restrict__ wo,
                  const float* __restrict__ bo, float* __restrict__ out)
{
    __shared__ __align__(16) bf16 As[2][BM * LDA];
    __shared__ __align__(16) bf16 Bs[2][BN * LDA];
    int tid = threadIdx.x, lane = tid & 31, wid = tid >> 5;
    int wm = wid >> 1, wn = wid & 1;
    int m0 = blockIdx.x * BM;
    int n0 = blockIdx.y * BN;
    int bt = blockIdx.z;
    const bf16* ab = attn + (size_t)bt * CH * HWN;

    auto stageA = [&](int k0, int b) {
#pragma unroll
        for (int j = 0; j < 2; ++j) {
            int c = j * NTHREADS + tid;
            int mr = c >> 2, kq = c & 3;
            const float* s = &wo[(size_t)(m0 + mr) * KCONV + k0 + kq * 8];
            float4 f0 = *(const float4*)s, f1 = *(const float4*)(s + 4);
            v8bf h = {(bf16)f0.x, (bf16)f0.y, (bf16)f0.z, (bf16)f0.w,
                      (bf16)f1.x, (bf16)f1.y, (bf16)f1.z, (bf16)f1.w};
            *(v8bf*)&As[b][mr * LDA + kq * 8] = h;
        }
    };
    // implicit im2col, transposed into [col][k] via octet gather + 16B store
    auto stageB = [&](int k0, int b) {
#pragma unroll
        for (int j = 0; j < 2; ++j) {
            int c = j * NTHREADS + tid;
            int kq = c >> 7, col = c & 127;
            int hw = n0 + col;
            int yy0 = hw >> 6, xx0 = hw & 63;
            bf16 t[8];
#pragma unroll
            for (int e = 0; e < 8; ++e) {
                int k  = k0 + kq * 8 + e;
                int ci = k / 9, jj = k - ci * 9;
                int yy = yy0 + jj / 3 - 1, xx = xx0 + jj % 3 - 1;
                bf16 v = (bf16)0.0f;
                if ((unsigned)yy < 64u && (unsigned)xx < 64u)
                    v = ab[(size_t)ci * HWN + yy * WW + xx];
                t[e] = v;
            }
            v8bf h = {t[0], t[1], t[2], t[3], t[4], t[5], t[6], t[7]};
            *(v8bf*)&Bs[b][col * LDA + kq * 8] = h;
        }
    };

    v8f acc[2][4] = {};
    const int steps = KCONV / BK;
    stageA(0, 0); stageB(0, 0);
    for (int i = 0; i < steps; ++i) {
        __syncthreads();
        if (i + 1 < steps) { stageA((i + 1) * BK, (i + 1) & 1); stageB((i + 1) * BK, (i + 1) & 1); }
        int cb = i & 1;
        COMPUTE_STEP(As[cb], Bs[cb]);
    }

    int cn = lane & 15, mrow0 = (lane < 16) ? 0 : 8;
#pragma unroll
    for (int im = 0; im < 2; ++im)
#pragma unroll
    for (int jn = 0; jn < 4; ++jn) {
        int hw = n0 + wn * 64 + jn * 16 + cn;
#pragma unroll
        for (int r = 0; r < 8; ++r) {
            int o = m0 + wm * 32 + im * 16 + mrow0 + r;
            float v = acc[im][jn][r] + bo[o];
            v = (v > 0.f) ? v : 0.2f * v;
            out[((size_t)bt * CH + o) * HWN + hw] = v;
        }
    }
}

// =====================================================================
extern "C" void kernel_launch(void* const* d_in, const int* in_sizes, int n_in,
                              void* d_out, int out_size, void* d_ws, size_t ws_size,
                              hipStream_t stream)
{
    (void)in_sizes; (void)n_in; (void)out_size; (void)ws_size;
    const float* x  = (const float*)d_in[0];
    // d_in[1] (mask m) is mathematically dead in the reference.
    const float* wq = (const float*)d_in[2];
    const float* bq = (const float*)d_in[3];
    const float* wk = (const float*)d_in[4];
    const float* bk = (const float*)d_in[5];
    const float* wv = (const float*)d_in[6];
    const float* bv = (const float*)d_in[7];
    const float* wo = (const float*)d_in[8];
    const float* bo = (const float*)d_in[9];
    float* out = (float*)d_out;

    const size_t packElems = (size_t)3 * BB * NDPB;
    bf16* qpack  = (bf16*)d_ws;
    bf16* kpack  = qpack + packElems;
    bf16* vpackT = kpack + packElems;
    bf16* attn   = vpackT + packElems;
    char* tail   = (char*)(attn + (size_t)BT * CH * HWN);
    float* scores = (float*)tail;
    bf16*  pmat   = (bf16*)(tail + (size_t)BB * 2048 * 2048 * 4);

    dim3 blk(NTHREADS);

    qkv_gemm_pack<<<dim3(CH / BM, HWN / BN, BT * 3), blk, 0, stream>>>(
        x, wq, bq, wk, bk, wv, bv, qpack, kpack, vpackT);

    const int NwinA[3]    = {128, 512, 2048};
    const int DdA[3]      = {65536, 16384, 4096};
    const int nchunksA[3] = {16, 2, 1};          // split-K for small-N / huge-K scales
    for (int s = 0; s < 3; ++s) {
        int Nwin = NwinA[s], Dd = DdA[s], nchunks = nchunksA[s];
        int KC = Dd / nchunks;
        float sfac = 1.0f / sqrtf((float)Dd);
        if (nchunks > 1) {
            int nz = BB * Nwin * Nwin;
            zero_f32<<<dim3((nz + 255) / 256), dim3(256), 0, stream>>>(scores, nz);
        }
        scores_gemm<<<dim3(Nwin / BM, Nwin / BN, BB * nchunks), blk, 0, stream>>>(
            qpack, kpack, scores, s, Nwin, Dd, KC, nchunks, sfac);
        softmax_rows<<<dim3(BB * Nwin), dim3(256), 0, stream>>>(scores, pmat, Nwin);
        if (s == 0)
            y_gemm_unpack<0><<<dim3(NwinA[0] / BM, DdA[0] / BN, BB), blk, 0, stream>>>(pmat, vpackT, attn);
        else if (s == 1)
            y_gemm_unpack<1><<<dim3(NwinA[1] / BM, DdA[1] / BN, BB), blk, 0, stream>>>(pmat, vpackT, attn);
        else
            y_gemm_unpack<2><<<dim3(NwinA[2] / BM, DdA[2] / BN, BB), blk, 0, stream>>>(pmat, vpackT, attn);
    }

    conv3x3_gemm<<<dim3(CH / BM, HWN / BN, BT), blk, 0, stream>>>(attn, wo, bo, out);
}
